// SpectralConv2dRealFreq_52716428591410
// MI455X (gfx1250) — matrix operand order repaired and checked by reading, not compiled, not run on hardware
//
#include <hip/hip_runtime.h>
#include <hip/hip_bf16.h>

// ---------------------------------------------------------------------------
// SpectralConv2dRealFreq on MI455X (gfx1250): 5-stage GEMM pipeline on
// v_wmma_f32_16x16x32_bf16 with all operands stored in fragment-swizzled
// order so every lane's fragment is one contiguous 32B load.
// ---------------------------------------------------------------------------

typedef __bf16 bf16_t;
typedef __attribute__((ext_vector_type(16))) __bf16 v16bf;
typedef __attribute__((ext_vector_type(8)))  float  v8f;

#define B_   16
#define C_   64
#define H_   128
#define W_   128
#define MX_  33
#define PI_F 3.14159265358979f

// scratch layout (bf16 element offsets; every region is 512-element aligned)
constexpr long OFF_WF  = 0;                                   // B [128][80]  -> 20 tiles
constexpr long OFF_AF1 = OFF_WF  + 20L * 512;                 // A [48][256]  -> 24 tiles
constexpr long OFF_AF2 = OFF_AF1 + 24L * 512;
constexpr long OFF_AI1 = OFF_AF2 + 24L * 512;                 // A [128][96]  -> 24 tiles
constexpr long OFF_AI2 = OFF_AI1 + 24L * 512;
constexpr long OFF_BI  = OFF_AI2 + 24L * 512;                 // B [96][128]  -> 24 tiles
constexpr long OFF_X1  = OFF_BI  + 24L * 512;                 // per(b,c) B [256][48] = 24 tiles
constexpr long OFF_XF  = OFF_X1  + 1024L * 24 * 512;          // per(i,j) A [16][128] = 4 tiles
constexpr long OFF_KP  = OFF_XF  + 1089L * 4 * 512;           // per(b,o) B [96][48]  = 9 tiles
constexpr long OFF_TP  = OFF_KP  + 1024L * 9 * 512;           // per(b,o) A [128][96] = 24 tiles
constexpr long OFF_WB1 = OFF_TP  + 1024L * 24 * 512;          // per(i,j) B [128][64] = 16 tiles
constexpr long OFF_WB2 = OFF_WB1 + 1089L * 16 * 512;

// ---------------------------------------------------------------------------
// helpers
// ---------------------------------------------------------------------------
__device__ __forceinline__ bf16_t f2bf(float f) {
  unsigned int u = __builtin_bit_cast(unsigned int, f);
  unsigned int r = (u + 0x7FFFu + ((u >> 16) & 1u)) >> 16;
  return __builtin_bit_cast(bf16_t, (unsigned short)r);
}

// swizzled position of logical (k,n) in a B table [K][N], tiles 32x16
__device__ __forceinline__ int bswz(int k, int n, int ntiles) {
  int kt = k >> 5, kl = k & 31, nt = n >> 4;
  int lane = ((kl & 8) ? 16 : 0) + (n & 15);
  return (((kt * ntiles + nt) * 32 + lane) << 4) + ((kl & 16) ? 8 : 0) + (kl & 7);
}
// swizzled position of logical (m,k) in an A table [M][K], tiles 16x32
__device__ __forceinline__ int aswz(int m, int k, int ktiles) {
  int mt = m >> 4, kl = k & 31, kt = k >> 5;
  int lane = (m & 15) + ((kl & 8) ? 16 : 0);
  int e = ((kl & 16) ? 8 : 0) + (kl & 7);
  return (((mt * ktiles + kt) * 32 + lane) << 4) + e;
}
// fragment = one contiguous 32B vector per lane
__device__ __forceinline__ v16bf load_frag(const bf16_t* base, int tile, int lane) {
  return *(const v16bf*)(base + (((long)tile * 32 + lane) << 4));
}

__device__ __forceinline__ v8f wmma_bf16(v16bf a, v16bf b, v8f c) {
  return __builtin_amdgcn_wmma_f32_16x16x32_bf16(false, a, false, b,
                                                 (short)0, c, false, false);
}

__device__ __forceinline__ float kfreq(int j) {
  return (j <= 16) ? (float)j : (float)(j - MX_);
}

// ---------------------------------------------------------------------------
// kernel 0: phase tables, written pre-swizzled
// ---------------------------------------------------------------------------
__global__ __launch_bounds__(256) void k_phases(bf16_t* ws) {
  int t = blockIdx.x * blockDim.x + threadIdx.x;
  const float inv = 0.0883883476483184f; // 1/sqrt(128)

  if (t < 128 * 80) {                 // WF: B table, K=m (128), N=col (80)
    int m = t / 80, col = t % 80;
    float v = 0.f;
    if (col < 66) {
      int j = (col < 33) ? col : col - 33;
      float ang = -2.f * PI_F * kfreq(j) * (float)m / 128.f;
      v = ((col < 33) ? cosf(ang) : sinf(ang)) * inv;
    }
    ws[OFF_WF + bswz(m, col, 5)] = f2bf(v);
  }
  if (t < 48 * 256) {                 // AF1/AF2: A tables [48][256]
    int i = t / 256, q = t % 256;
    float v1 = 0.f, v2 = 0.f;
    if (i < 33) {
      int n = q & 127;
      float ang = -2.f * PI_F * kfreq(i) * (float)n / 128.f;
      float hr = cosf(ang) * inv, hi = sinf(ang) * inv;
      if (q < 128) { v1 = hr;  v2 = hi; }
      else         { v1 = -hi; v2 = hr; }
    }
    int p = aswz(i, q, 8);
    ws[OFF_AF1 + p] = f2bf(v1);
    ws[OFF_AF2 + p] = f2bf(v2);
  }
  if (t < 128 * 96) {                 // AI1/AI2: A tables [128][96]
    int n = t / 96, k = t % 96;
    float v1 = 0.f, v2 = 0.f;
    if (k < 66) {
      int i = (k < 33) ? k : k - 33;
      float ang = 2.f * PI_F * kfreq(i) * (float)n / 128.f;
      float pr = cosf(ang) * inv, pi = sinf(ang) * inv;
      if (k < 33) { v1 = pr;  v2 = pi; }
      else        { v1 = -pi; v2 = pr; }
    }
    int p = aswz(n, k, 3);
    ws[OFF_AI1 + p] = f2bf(v1);
    ws[OFF_AI2 + p] = f2bf(v2);
  }
  if (t < 96 * 128) {                 // BI: B table [96][128]
    int k = t / 128, m = t % 128;
    float v = 0.f;
    if (k < 66) {
      int j = (k < 33) ? k : k - 33;
      float ang = 2.f * PI_F * kfreq(j) * (float)m / 128.f;
      v = ((k < 33) ? cosf(ang) : -sinf(ang)) * inv;
    }
    ws[OFF_BI + bswz(k, m, 8)] = f2bf(v);
  }
}

// ---------------------------------------------------------------------------
// kernel 0b: weight transpose fp32 [c][o][i][j] -> bf16 swizzled per-(i,j)
// WB1 = [Rr ; -Ri], WB2 = [Ri ; Rr]  (K = 2*64, N = 64)
// ---------------------------------------------------------------------------
__global__ __launch_bounds__(256) void k_wprep(const float* __restrict__ wr,
                                               const float* __restrict__ wi,
                                               bf16_t* __restrict__ ws) {
  long t = (long)blockIdx.x * blockDim.x + threadIdx.x;
  if (t >= (long)C_ * C_ * MX_ * MX_) return;
  int j = (int)(t % MX_);
  int i = (int)((t / MX_) % MX_);
  int o = (int)((t / (MX_ * MX_)) % C_);
  int c = (int)(t / ((long)MX_ * MX_ * C_));
  float rr = wr[t], ri = wi[t];
  long base1 = OFF_WB1 + (long)(i * MX_ + j) * (16 * 512);
  long base2 = OFF_WB2 + (long)(i * MX_ + j) * (16 * 512);
  int pLo = bswz(c, o, 4), pHi = bswz(64 + c, o, 4);
  ws[base1 + pLo] = f2bf(rr);
  ws[base1 + pHi] = f2bf(-ri);
  ws[base2 + pLo] = f2bf(ri);
  ws[base2 + pHi] = f2bf(rr);
}

// ---------------------------------------------------------------------------
// stage 1: [131072 x 128] x [128 x 66] -> X1 (B-swizzled per (b,c))
// ---------------------------------------------------------------------------
__global__ __launch_bounds__(128) void k_stage1(const float* __restrict__ x,
                                                bf16_t* __restrict__ ws) {
  __shared__ __align__(32) bf16_t Asm[4][4 * 512];   // per-wave A, swizzled
  int wave = threadIdx.x >> 5, lane = threadIdx.x & 31;
  long mtile = (long)blockIdx.x * 4 + wave;
  long row0 = mtile * 16;

  const float* xa = x + row0 * 128;
  __builtin_prefetch(xa + 16 * 128, 0, 1);           // global_prefetch_b8
  bf16_t* As = Asm[wave];
  // stage 16x128 fp32 -> bf16 swizzled: float4 coalesced reads, b64 LDS writes
  const float4* xv = (const float4*)xa;
#pragma unroll
  for (int it = 0; it < 16; ++it) {
    int q = lane + it * 32;          // float4 chunk id (512 total)
    int m = q >> 5, kc = (q & 31) << 2;
    float4 f = xv[q];
    int kt = kc >> 5, kl = kc & 31;
    int lw = (m & 15) + ((kl & 8) ? 16 : 0);
    int e  = ((kl & 16) ? 8 : 0) + (kl & 7);
    bf16_t* d = As + ((kt * 32 + lw) << 4) + e;
    d[0] = f2bf(f.x); d[1] = f2bf(f.y); d[2] = f2bf(f.z); d[3] = f2bf(f.w);
  }

  const bf16_t* WF = ws + OFF_WF;
  v8f acc[5] = {};
#pragma unroll
  for (int kt = 0; kt < 4; ++kt) {
    v16bf a = load_frag(As, kt, lane);
#pragma unroll
    for (int nt = 0; nt < 5; ++nt)
      acc[nt] = wmma_bf16(a, load_frag(WF, kt * 5 + nt, lane), acc[nt]);
  }
  bf16_t* X1 = ws + OFF_X1;
  int n0 = lane & 15, mb = (lane & 16) ? 8 : 0;
#pragma unroll
  for (int nt = 0; nt < 5; ++nt) {
    int j = nt * 16 + n0;
    if (j >= 66) continue;
    int slab = (j < 33) ? 0 : 1;
    int jj = (j < 33) ? j : j - 33;
#pragma unroll
    for (int r = 0; r < 8; ++r) {
      long row = row0 + mb + r;
      long bc = row >> 7;
      int  n  = (int)(row & 127);
      X1[bc * (24 * 512) + bswz(slab * 128 + n, jj, 3)] = f2bf(acc[nt][r]);
    }
  }
}

// ---------------------------------------------------------------------------
// stage 2: per (b,c): [48x256] x [256x48] x2 comps -> XF (A-swizzled per (i,j))
// ---------------------------------------------------------------------------
__global__ __launch_bounds__(128) void k_stage2(bf16_t* __restrict__ ws) {
  __shared__ __align__(32) bf16_t Bs[24 * 512];      // 24KB, already swizzled
  int bc = blockIdx.x;
  {
    const uint4* s = (const uint4*)(ws + OFF_X1 + (long)bc * (24 * 512));
    uint4* d = (uint4*)Bs;
    for (int e = threadIdx.x; e < 24 * 512 / 8; e += 128) d[e] = s[e];
  }
  __syncthreads();

  int wave = threadIdx.x >> 5, lane = threadIdx.x & 31;
  int b = bc >> 6, c = bc & 63;
  bf16_t* XF = ws + OFF_XF;

  for (int job = wave; job < 18; job += 4) {         // comp(2) x mt(3) x nt(3)
    int comp = job / 9, mt = (job % 9) / 3, nt = job % 3;
    const bf16_t* A = ws + (comp ? OFF_AF2 : OFF_AF1);
    v8f acc = {};
#pragma unroll
    for (int kt = 0; kt < 8; ++kt)
      acc = wmma_bf16(load_frag(A, mt * 8 + kt, lane),
                      load_frag(Bs, kt * 3 + nt, lane), acc);
    int n0 = lane & 15, mb = (lane & 16) ? 8 : 0;
    int j = nt * 16 + n0;
    int p0 = aswz(b, comp * 64 + c, 4);              // constant per job
    if (j < 33) {
#pragma unroll
      for (int r = 0; r < 8; ++r) {
        int i = mt * 16 + mb + r;
        if (i < 33)
          XF[(long)(i * MX_ + j) * (4 * 512) + p0] = f2bf(acc[r]);
      }
    }
  }
}

// ---------------------------------------------------------------------------
// stage 3: per (i,j): [16x128] x [128x64] x2 comps, all-swizzled global, no LDS
// -> KP (B-swizzled per (b,o))
// ---------------------------------------------------------------------------
__global__ __launch_bounds__(128) void k_stage3(bf16_t* __restrict__ ws) {
  int ij = blockIdx.x;
  int wave = threadIdx.x >> 5, lane = threadIdx.x & 31;
  int i = ij / MX_, j = ij % MX_;
  const bf16_t* Aij = ws + OFF_XF + (long)ij * (4 * 512);
  bf16_t* KP = ws + OFF_KP;
  int nt = wave;                                     // 4 n-tiles over 4 waves
  for (int comp = 0; comp < 2; ++comp) {
    const bf16_t* Bij = ws + (comp ? OFF_WB2 : OFF_WB1) + (long)ij * (16 * 512);
    v8f acc = {};
#pragma unroll
    for (int kt = 0; kt < 4; ++kt)
      acc = wmma_bf16(load_frag(Aij, kt, lane),
                      load_frag(Bij, kt * 4 + nt, lane), acc);
    int n0 = lane & 15, mb = (lane & 16) ? 8 : 0;
    int o = nt * 16 + n0;
    int rowk = comp ? (33 + i) : i;
    int pc = bswz(rowk, j, 3);                       // constant per comp
#pragma unroll
    for (int r = 0; r < 8; ++r) {
      int bb = mb + r;
      KP[(long)(bb * 64 + o) * (9 * 512) + pc] = f2bf(acc[r]);
    }
  }
}

// ---------------------------------------------------------------------------
// stage 4: per (b,o): [128x96] x [96x48] x2 comps -> TP (A-swizzled per (b,o))
// ---------------------------------------------------------------------------
__global__ __launch_bounds__(128) void k_stage4(bf16_t* __restrict__ ws) {
  __shared__ __align__(32) bf16_t Bs[9 * 512];       // 9KB, swizzled
  int bo = blockIdx.x;
  {
    const uint4* s = (const uint4*)(ws + OFF_KP + (long)bo * (9 * 512));
    uint4* d = (uint4*)Bs;
    for (int e = threadIdx.x; e < 9 * 512 / 8; e += 128) d[e] = s[e];
  }
  __syncthreads();

  int wave = threadIdx.x >> 5, lane = threadIdx.x & 31;
  bf16_t* TP = ws + OFF_TP + (long)bo * (24 * 512);

  for (int job = wave; job < 48; job += 4) {         // comp(2) x mt(8) x nt(3)
    int comp = job / 24, mt = (job % 24) / 3, nt = job % 3;
    const bf16_t* A = ws + (comp ? OFF_AI2 : OFF_AI1);
    v8f acc = {};
#pragma unroll
    for (int kt = 0; kt < 3; ++kt)
      acc = wmma_bf16(load_frag(A, mt * 3 + kt, lane),
                      load_frag(Bs, kt * 3 + nt, lane), acc);
    int n0 = lane & 15, mb = (lane & 16) ? 8 : 0;
    int j = nt * 16 + n0;
    if (j < 33) {
#pragma unroll
      for (int r = 0; r < 8; ++r) {
        int n = mt * 16 + mb + r;
        TP[aswz(n, comp * 33 + j, 3)] = f2bf(acc[r]);
      }
    }
  }
}

// ---------------------------------------------------------------------------
// stage 5: [131072 x 96] x [96 x 128] -> out fp32
// ---------------------------------------------------------------------------
__global__ __launch_bounds__(128) void k_stage5(const bf16_t* __restrict__ ws,
                                                float* __restrict__ out) {
  int wave = threadIdx.x >> 5, lane = threadIdx.x & 31;
  long mtile = (long)blockIdx.x * 4 + wave;
  long bo = mtile >> 3;
  int mloc = (int)(mtile & 7);
  const bf16_t* Abo = ws + OFF_TP + bo * (24 * 512);
  const bf16_t* BI = ws + OFF_BI;
  __builtin_prefetch(Abo + (mloc * 3) * 512, 0, 1);  // global_prefetch_b8
  float* orow = out + mtile * 16 * 128;

  v8f acc[8] = {};
#pragma unroll
  for (int kt = 0; kt < 3; ++kt) {
    v16bf a = load_frag(Abo, mloc * 3 + kt, lane);
#pragma unroll
    for (int nt = 0; nt < 8; ++nt)
      acc[nt] = wmma_bf16(a, load_frag(BI, kt * 8 + nt, lane), acc[nt]);
  }
  int n0 = lane & 15, mb = (lane & 16) ? 8 : 0;
#pragma unroll
  for (int nt = 0; nt < 8; ++nt)
#pragma unroll
    for (int r = 0; r < 8; ++r)
      orow[(long)(mb + r) * 128 + nt * 16 + n0] = acc[nt][r];
}

// ---------------------------------------------------------------------------
extern "C" void kernel_launch(void* const* d_in, const int* in_sizes, int n_in,
                              void* d_out, int out_size, void* d_ws, size_t ws_size,
                              hipStream_t stream) {
  (void)in_sizes; (void)n_in; (void)out_size; (void)ws_size;
  const float* x  = (const float*)d_in[0];
  const float* wr = (const float*)d_in[1];
  const float* wi = (const float*)d_in[2];
  float* out = (float*)d_out;
  bf16_t* ws = (bf16_t*)d_ws;

  k_phases<<<48, 256, 0, stream>>>(ws);
  long wtot = (long)C_ * C_ * MX_ * MX_;
  k_wprep<<<(int)((wtot + 255) / 256), 256, 0, stream>>>(wr, wi, ws);
  k_stage1<<<2048, 128, 0, stream>>>(x, ws);
  k_stage2<<<1024, 128, 0, stream>>>(ws);
  k_stage3<<<MX_ * MX_, 128, 0, stream>>>(ws);
  k_stage4<<<1024, 128, 0, stream>>>(ws);
  k_stage5<<<2048, 128, 0, stream>>>(ws, out);
}
